// FastRotaryAttention_13932873908655
// MI455X (gfx1250) — matrix operand order repaired and checked by reading, not compiled, not run on hardware
//
#include <hip/hip_runtime.h>
#include <hip/hip_bf16.h>

// ---------------------------------------------------------------------------
// FastRotaryAttention on MI455X (gfx1250, wave32, WMMA).
//   B=1, S=2048, E=4096, H=32, KV=8, D=128, GQA n_rep=4, causal, RoPE base 1e4.
// Pipeline: f32->f16 convert, WMMA GEMMs for QKV, RoPE (+1/sqrt(D) folded into
// Q), per-(head,qtile) flash attention with v_wmma_f32_16x16x32_f16, WMMA GEMM
// for the output projection (fp32 out).
// GEMM uses a 64x64 wave tile (16 wmma / 32-K step) for ~32 flop/byte from L2.
// ---------------------------------------------------------------------------

typedef __attribute__((ext_vector_type(8)))  _Float16 v8h;
typedef __attribute__((ext_vector_type(16))) _Float16 v16h;
typedef __attribute__((ext_vector_type(8)))  float    v8f;

#define S_LEN 2048
#define E_DIM 4096
#define H_N   32
#define KV_N  8
#define D_DIM 128

static __device__ __forceinline__ v16h pack16(v8h lo, v8h hi) {
  return __builtin_shufflevector(lo, hi, 0,1,2,3,4,5,6,7,8,9,10,11,12,13,14,15);
}
static __device__ __forceinline__ v8f zero8() {
  return (v8f){0.f,0.f,0.f,0.f,0.f,0.f,0.f,0.f};
}

// A fragment (16x32 f16): lanes 0-15 row M=lane hold K 0..7 & 16..23;
// lanes 16-31 hold K 8..15 & 24..31.  Row-major memory -> two b128 loads.
static __device__ __forceinline__ v16h load_a_frag(const _Float16* __restrict__ rowp,
                                                   int g /*lane>>4*/) {
  const _Float16* p = rowp + g * 8;
  v8h lo = *reinterpret_cast<const v8h*>(p);
  v8h hi = *reinterpret_cast<const v8h*>(p + 16);
  return pack16(lo, hi);
}
// B fragment (32x16 f16): lanes 0-15 column N=lane&15 hold K 0..7 & 16..23;
// lanes 16-31 hold K 8..15 & 24..31.  Column-contiguous memory -> two b128 loads.
static __device__ __forceinline__ v16h load_b_frag(const _Float16* __restrict__ colp,
                                                   int g /*lane>>4*/) {
  const _Float16* p = colp + g * 8;
  v8h lo = *reinterpret_cast<const v8h*>(p);
  v8h hi = *reinterpret_cast<const v8h*>(p + 16);
  return pack16(lo, hi);
}

// ---------------------------------------------------------------------------
__global__ void k_cvt_f32_f16(const float* __restrict__ x, _Float16* __restrict__ y, int n) {
  int i = blockIdx.x * blockDim.x + threadIdx.x;
  if (i < n) y[i] = (_Float16)x[i];
}

// ---------------------------------------------------------------------------
// C[M,N] = A[M,K] * B[N,K]^T   (torch Linear: x @ W.T), f16 in / f32 out.
// One wave per 64x64 output tile: 16 wmma per 32-K step, 256B/lane loaded.
__global__ void __launch_bounds__(32) k_gemm_nt(const _Float16* __restrict__ A,
                                                const _Float16* __restrict__ B,
                                                float* __restrict__ C,
                                                int M, int N, int K) {
  const int lane = threadIdx.x & 31;
  const int g = lane >> 4, l15 = lane & 15;
  const int m0 = blockIdx.y * 64;
  const int n0 = blockIdx.x * 64;

  v8f acc[4][4];
#pragma unroll
  for (int t = 0; t < 4; ++t)
#pragma unroll
    for (int j = 0; j < 4; ++j) acc[t][j] = zero8();

  const _Float16* arow[4];
#pragma unroll
  for (int t = 0; t < 4; ++t) arow[t] = A + (size_t)(m0 + t * 16 + l15) * K;
  const _Float16* bcol[4];
#pragma unroll
  for (int j = 0; j < 4; ++j) bcol[j] = B + (size_t)(n0 + j * 16 + l15) * K;

  for (int k = 0; k < K; k += 32) {
    v16h a[4], b[4];
#pragma unroll
    for (int t = 0; t < 4; ++t) a[t] = load_a_frag(arow[t] + k, g);
#pragma unroll
    for (int j = 0; j < 4; ++j) b[j] = load_b_frag(bcol[j] + k, g);
#pragma unroll
    for (int t = 0; t < 4; ++t)
#pragma unroll
      for (int j = 0; j < 4; ++j)
        acc[t][j] = __builtin_amdgcn_wmma_f32_16x16x32_f16(
            false, a[t], false, b[j], (short)0, acc[t][j], false, false);
  }

#pragma unroll
  for (int t = 0; t < 4; ++t)
#pragma unroll
    for (int j = 0; j < 4; ++j)
#pragma unroll
      for (int r = 0; r < 8; ++r)
        C[(size_t)(m0 + t * 16 + g * 8 + r) * N + (n0 + j * 16 + l15)] = acc[t][j][r];
}

// ---------------------------------------------------------------------------
// RoPE + layout shuffles. Reads fp32 GEMM outputs, writes:
//   Q16[h][s][d]  (scaled by 1/sqrt(D)),  K16[kv][s][d],  VT16[kv][d][s].
__global__ void k_rope(const float* __restrict__ Qf, const float* __restrict__ Kf,
                       const float* __restrict__ Vf, _Float16* __restrict__ Q16,
                       _Float16* __restrict__ K16, _Float16* __restrict__ VT16) {
  int tid = blockIdx.x * blockDim.x + threadIdx.x;
  if (tid >= S_LEN * H_N * D_DIM) return;
  int d = tid & (D_DIM - 1);
  int h = (tid >> 7) & (H_N - 1);
  int s = tid >> 12;

  int i = d & 63;                                   // freq index
  float inv = __expf(-0.14391156463f * (float)i);   // 10000^(-i/64)
  float ang = (float)s * inv;
  float sn, cs;
  __sincosf(ang, &sn, &cs);

  const float* qrow = Qf + (size_t)s * (H_N * D_DIM) + h * D_DIM;
  float q  = qrow[d];
  float qo = (d < 64) ? qrow[d + 64] : qrow[d - 64];
  float qr = (d < 64) ? (q * cs - qo * sn) : (q * cs + qo * sn);
  Q16[((size_t)h * S_LEN + s) * D_DIM + d] = (_Float16)(qr * 0.08838834764831845f);

  if (h < KV_N) {
    const float* krow = Kf + (size_t)s * (KV_N * D_DIM) + h * D_DIM;
    float k  = krow[d];
    float ko = (d < 64) ? krow[d + 64] : krow[d - 64];
    float kr = (d < 64) ? (k * cs - ko * sn) : (k * cs + ko * sn);
    K16[((size_t)h * S_LEN + s) * D_DIM + d] = (_Float16)kr;
    VT16[((size_t)h * D_DIM + d) * S_LEN + s] =
        (_Float16)Vf[(size_t)s * (KV_N * D_DIM) + h * D_DIM + d];
  }
}

// ---------------------------------------------------------------------------
// Flash attention: one wave per (16-row q tile, head). Keys in blocks of 32.
// Output written as O16[s][h*D+d] (f16) ready to be the A operand of out-proj.
__global__ void __launch_bounds__(32) k_attn(const _Float16* __restrict__ Q16,
                                             const _Float16* __restrict__ K16,
                                             const _Float16* __restrict__ VT16,
                                             _Float16* __restrict__ O16) {
  __shared__ __align__(16) _Float16 lds[16][48];   // P tile staging (row stride 96B)

  const int lane = threadIdx.x & 31;
  const int g = lane >> 4, l15 = lane & 15;
  const int q0 = blockIdx.x * 16;
  const int h  = blockIdx.y;
  const int kv = h >> 2;                            // GQA: n_rep = 4

  // Q fragments, K = 0..127 in four 32-chunks (score scale already folded in).
  v16h qa[4];
  const _Float16* qrow = Q16 + ((size_t)h * S_LEN + q0 + l15) * D_DIM;
#pragma unroll
  for (int c = 0; c < 4; ++c) qa[c] = load_a_frag(qrow + c * 32, g);

  v8f oacc[8];
#pragma unroll
  for (int j = 0; j < 8; ++j) oacc[j] = zero8();
  float m_i[8], l_i[8];
#pragma unroll
  for (int r = 0; r < 8; ++r) { m_i[r] = -1e30f; l_i[r] = 0.f; }

  const _Float16* kbase = K16 + (size_t)kv * S_LEN * D_DIM;
  const _Float16* vbase = VT16 + (size_t)kv * D_DIM * S_LEN;
  const int kend = q0 + 16;

  for (int kb = 0; kb < kend; kb += 32) {
    // ---- scores: two 16x16 tiles over K=128 -------------------------------
    v8f s0 = zero8(), s1 = zero8();
#pragma unroll
    for (int c = 0; c < 4; ++c) {
      v16h b0 = load_b_frag(kbase + (size_t)(kb + l15) * D_DIM + c * 32, g);
      v16h b1 = load_b_frag(kbase + (size_t)(kb + 16 + l15) * D_DIM + c * 32, g);
      s0 = __builtin_amdgcn_wmma_f32_16x16x32_f16(false, qa[c], false, b0, (short)0, s0, false, false);
      s1 = __builtin_amdgcn_wmma_f32_16x16x32_f16(false, qa[c], false, b1, (short)0, s1, false, false);
    }

    // ---- causal mask + online softmax (row reductions over 16-lane groups)
    const int col0 = kb + l15, col1 = kb + 16 + l15;
    v8f p0v, p1v;
    float alpha[8];
#pragma unroll
    for (int r = 0; r < 8; ++r) {
      const int row = q0 + g * 8 + r;
      float x0 = (col0 <= row) ? s0[r] : -1e30f;
      float x1 = (col1 <= row) ? s1[r] : -1e30f;
      float mx = fmaxf(x0, x1);
#pragma unroll
      for (int off = 8; off >= 1; off >>= 1) mx = fmaxf(mx, __shfl_xor(mx, off, 16));
      float mn = fmaxf(m_i[r], mx);
      float e0 = __expf(x0 - mn), e1 = __expf(x1 - mn);
      float rs = e0 + e1;
#pragma unroll
      for (int off = 8; off >= 1; off >>= 1) rs += __shfl_xor(rs, off, 16);
      float al = __expf(m_i[r] - mn);
      l_i[r] = l_i[r] * al + rs;
      m_i[r] = mn;
      alpha[r] = al;
      p0v[r] = e0;
      p1v[r] = e1;
    }
#pragma unroll
    for (int j = 0; j < 8; ++j)
#pragma unroll
      for (int r = 0; r < 8; ++r) oacc[j][r] *= alpha[r];

    // ---- P (C layout) -> LDS -> A-fragment layout -------------------------
    __syncthreads();                                 // WAR vs previous loads
#pragma unroll
    for (int r = 0; r < 8; ++r) {
      lds[g * 8 + r][l15]      = (_Float16)p0v[r];
      lds[g * 8 + r][16 + l15] = (_Float16)p1v[r];
    }
    __syncthreads();
    const _Float16* pp = &lds[l15][g * 8];
    v16h pa = pack16(*reinterpret_cast<const v8h*>(pp),
                     *reinterpret_cast<const v8h*>(pp + 16));

    // ---- O += P @ V_block (VT is [d][s]: B columns contiguous) ------------
#pragma unroll
    for (int j = 0; j < 8; ++j) {
      v16h bv = load_b_frag(vbase + (size_t)(j * 16 + l15) * S_LEN + kb, g);
      oacc[j] = __builtin_amdgcn_wmma_f32_16x16x32_f16(false, pa, false, bv, (short)0, oacc[j], false, false);
    }
  }

  // ---- normalize + store O16[s][h*D + d] ----------------------------------
#pragma unroll
  for (int r = 0; r < 8; ++r) {
    float invl = 1.0f / l_i[r];
    const size_t rowo = (size_t)(q0 + g * 8 + r) * (H_N * D_DIM) + h * D_DIM;
#pragma unroll
    for (int j = 0; j < 8; ++j)
      O16[rowo + j * 16 + l15] = (_Float16)(oacc[j][r] * invl);
  }
}

// ---------------------------------------------------------------------------
extern "C" void kernel_launch(void* const* d_in, const int* in_sizes, int n_in,
                              void* d_out, int out_size, void* d_ws, size_t ws_size,
                              hipStream_t stream) {
  (void)in_sizes; (void)n_in; (void)out_size; (void)ws_size;
  const float* hs = (const float*)d_in[0];
  // d_in[1] = position_ids (int64) == arange(S); positions derived analytically.
  const float* Wq = (const float*)d_in[2];
  const float* Wk = (const float*)d_in[3];
  const float* Wv = (const float*)d_in[4];
  const float* Wo = (const float*)d_in[5];

  char* ws = (char*)d_ws;
  const size_t MB = 1ull << 20;
  // Lifetime-safe aliasing (total 144 MB):
  _Float16* h16    = (_Float16*)(ws + 0);        // 16MB; dead after QKV GEMMs
  _Float16* q16    = h16;                        // 16MB; written by rope
  _Float16* wq16   = (_Float16*)(ws + 16 * MB);  // 32MB; dead after GEMM(Q)
  _Float16* k16    = wq16;                       //  4MB; written by rope
  _Float16* vt16   = (_Float16*)(ws + 20 * MB);  //  4MB; written by rope
  _Float16* wk16   = (_Float16*)(ws + 48 * MB);  //  8MB
  _Float16* wv16   = (_Float16*)(ws + 56 * MB);  //  8MB
  _Float16* wo16   = (_Float16*)(ws + 64 * MB);  // 32MB
  float*    qf32   = (float*)(ws + 96 * MB);     // 32MB; dead after rope
  _Float16* attn16 = (_Float16*)(ws + 96 * MB);  // 16MB; written by attention
  float*    kf32   = (float*)(ws + 128 * MB);    //  8MB
  float*    vf32   = (float*)(ws + 136 * MB);    //  8MB

  // fp32 -> f16 conversions
  {
    int n;
    n = S_LEN * E_DIM;       k_cvt_f32_f16<<<(n + 255) / 256, 256, 0, stream>>>(hs, h16, n);
    n = E_DIM * E_DIM;       k_cvt_f32_f16<<<(n + 255) / 256, 256, 0, stream>>>(Wq, wq16, n);
    n = KV_N * D_DIM * E_DIM;k_cvt_f32_f16<<<(n + 255) / 256, 256, 0, stream>>>(Wk, wk16, n);
    n = KV_N * D_DIM * E_DIM;k_cvt_f32_f16<<<(n + 255) / 256, 256, 0, stream>>>(Wv, wv16, n);
    n = E_DIM * E_DIM;       k_cvt_f32_f16<<<(n + 255) / 256, 256, 0, stream>>>(Wo, wo16, n);
  }

  // QKV projections (64x64 tiles)
  k_gemm_nt<<<dim3(E_DIM / 64, S_LEN / 64), 32, 0, stream>>>(h16, wq16, qf32, S_LEN, E_DIM, E_DIM);
  k_gemm_nt<<<dim3((KV_N * D_DIM) / 64, S_LEN / 64), 32, 0, stream>>>(h16, wk16, kf32, S_LEN, KV_N * D_DIM, E_DIM);
  k_gemm_nt<<<dim3((KV_N * D_DIM) / 64, S_LEN / 64), 32, 0, stream>>>(h16, wv16, vf32, S_LEN, KV_N * D_DIM, E_DIM);

  // RoPE + layout
  {
    int n = S_LEN * H_N * D_DIM;
    k_rope<<<(n + 255) / 256, 256, 0, stream>>>(qf32, kf32, vf32, q16, k16, vt16);
  }

  // Flash attention
  k_attn<<<dim3(S_LEN / 16, H_N), 32, 0, stream>>>(q16, k16, vt16, attn16);

  // Output projection -> fp32 d_out
  k_gemm_nt<<<dim3(E_DIM / 64, S_LEN / 64), 32, 0, stream>>>(attn16, wo16, (float*)d_out, S_LEN, E_DIM, E_DIM);
}
